// myModel_34608846471950
// MI455X (gfx1250) — matrix operand order repaired and checked by reading, not compiled, not run on hardware
//
#include <hip/hip_runtime.h>
#include <hip/hip_bf16.h>

#define DIMV   128
#define RDIM   256
#define TSTEPS 8

typedef __attribute__((ext_vector_type(16))) __bf16 v16bf;
typedef __attribute__((ext_vector_type(8)))  __bf16 v8bf;
typedef __attribute__((ext_vector_type(8)))  float  v8f;
typedef __attribute__((ext_vector_type(4)))  unsigned int v4u;
typedef __attribute__((ext_vector_type(4)))  int    v4i;
typedef __attribute__((ext_vector_type(8)))  int    v8i;

__device__ __forceinline__ v8f wmma_bf16(v16bf a, v16bf b, v8f c) {
    // (neg_a, A, neg_b, B, c_mod, C, reuse_a, reuse_b)
    return __builtin_amdgcn_wmma_f32_16x16x32_bf16(false, a, false, b, (short)0, c,
                                                   false, false);
}

// ---- Tensor Data Mover: DMA a 2-D bf16 tile (whole matrix) from global to LDS.
// D# layout per CDNA5 ISA 8.3-8.5: group0 = {count, lds_addr, global_addr, type=2},
// group1 = {data_size=2B, tensor_dim0/1, tile_dim0/1, tensor_dim0_stride}, groups
// 2/3/4 zero (2-D tensor). 6-arg builtin form (clang-23 / amdgpu-toolchain lane).
__device__ __forceinline__ void tdm_load_2d_bf16(unsigned int lds_off,
                                                 const void* gptr,
                                                 unsigned int dim0,     // elems/row
                                                 unsigned int dim1,     // rows
                                                 unsigned int stride0)  // elems
{
    unsigned long long ga = (unsigned long long)(uintptr_t)gptr;
    v4u g0;
    g0.x = 1u;                                           // count=1, user D#
    g0.y = lds_off;                                      // lds_addr (bytes)
    g0.z = (unsigned int)(ga & 0xFFFFFFFFu);             // global_addr[95:64]
    g0.w = (unsigned int)((ga >> 32) & 0x01FFFFFFu) | 0x80000000u;  // addr hi + type=2
    v8i g1;
    g1[0] = (int)(1u << 16);                             // data_size=1 (2 bytes)
    g1[1] = (int)((dim0 & 0xFFFFu) << 16);               // tensor_dim0[15:0]
    g1[2] = (int)((dim0 >> 16) | ((dim1 & 0xFFFFu) << 16)); // dim0 hi | dim1 lo
    g1[3] = (int)((dim1 >> 16) | ((dim0 & 0xFFFFu) << 16)); // dim1 hi | tile_dim0
    g1[4] = (int)(dim1 & 0xFFFFu);                       // tile_dim1 (tile_dim2=0)
    g1[5] = (int)stride0;                                // tensor_dim0_stride lo
    g1[6] = 0;
    g1[7] = 0;
    v4i gz4 = {0, 0, 0, 0};
    v8i gz8 = {0, 0, 0, 0, 0, 0, 0, 0};
    __builtin_amdgcn_tensor_load_to_lds(g0, g1, gz4, gz4, gz8, 0);
}

__device__ __forceinline__ float selu_f(float x) {
    const float scale = 1.0507009873554805f;
    const float alpha = 1.6732632423543772f;
    return x > 0.f ? scale * x : scale * alpha * (__expf(x) - 1.f);
}

__device__ __forceinline__ float sigmoid_f(float x) {
    return 1.f / (1.f + __expf(-x));
}

// ---------------------------------------------------------------- utilities
__global__ void copy_f32_kernel(const float* __restrict__ src,
                                float* __restrict__ dst, int n) {
    int i = blockIdx.x * blockDim.x + threadIdx.x;
    if (i < n) dst[i] = src[i];
}

__global__ void zero_f32_kernel(float* __restrict__ dst, int n) {
    int i = blockIdx.x * blockDim.x + threadIdx.x;
    if (i < n) dst[i] = 0.f;
}

// state f32 -> bf16 shadow, and zero the aggregation buffer
__global__ void cvt_zero_kernel(const float* __restrict__ ls,
                                __bf16* __restrict__ lsbf,
                                float* __restrict__ agg, int n) {
    int i = blockIdx.x * blockDim.x + threadIdx.x;
    if (i < n) {
        lsbf[i] = (__bf16)ls[i];
        agg[i]  = 0.f;
    }
}

// transpose + convert weights to bf16 [N][K] layout
__global__ void prep_weights_kernel(const float* __restrict__ Wmsg,   // [2*DIMV, DIMV]
                                    const float* __restrict__ gk,     // [DIMV, 3*DIMV]
                                    const float* __restrict__ grk,    // [DIMV, 3*DIMV]
                                    __bf16* __restrict__ WmsgT,       // [DIMV][2*DIMV]
                                    __bf16* __restrict__ KT,          // [3*DIMV][DIMV]
                                    __bf16* __restrict__ RT) {        // [3*DIMV][DIMV]
    int idx = blockIdx.x * blockDim.x + threadIdx.x;
    if (idx < DIMV * 2 * DIMV) {
        int n = idx / (2 * DIMV), k = idx % (2 * DIMV);
        WmsgT[idx] = (__bf16)Wmsg[k * DIMV + n];
    }
    int idx2 = idx - DIMV * 2 * DIMV;
    if (idx2 >= 0 && idx2 < 3 * DIMV * DIMV) {
        int n = idx2 / DIMV, k = idx2 % DIMV;
        KT[idx2] = (__bf16)gk[k * (3 * DIMV) + n];
        RT[idx2] = (__bf16)grk[k * (3 * DIMV) + n];
    }
}

// ------------------------------------------------------- message + scatter
// One wave handles a 16-pair tile: C[16x128] = selu([ls[first]|ls[second]] @ Wmsg + b)
// then atomic scatter-add into agg[second]. W_msg (64KB bf16) is TDM-staged into
// LDS once per block and B-fragments are served from ds_load.
__global__ void msg_kernel(const __bf16* __restrict__ lsbf,   // [E][DIMV]
                           const __bf16* __restrict__ WmsgT,  // [DIMV][2*DIMV]
                           const float* __restrict__ bmsg,    // [DIMV]
                           const int* __restrict__ first,
                           const int* __restrict__ second,
                           float* __restrict__ agg,           // [E][DIMV]
                           int P) {
    __shared__ __bf16 sW[DIMV * 2 * DIMV];                    // 64 KB

    if (threadIdx.x < 32) {
        tdm_load_2d_bf16((unsigned int)(uintptr_t)&sW[0], WmsgT,
                         2 * DIMV, DIMV, 2 * DIMV);
        __builtin_amdgcn_s_wait_tensorcnt(0);
    }
    __syncthreads();

    const int lane  = threadIdx.x & 31;
    const int wave  = threadIdx.x >> 5;
    const int tile  = blockIdx.x * (blockDim.x >> 5) + wave;
    const int pbase = tile * 16;
    if (pbase + 16 > P) return;

    const int mrow = lane & 15;
    const int half = lane >> 4;

    const int rowF = first[pbase + mrow];
    const int rowS = second[pbase + mrow];

    // Preload the 8 A fragments (K = 2*DIMV = 8 chunks of 32).
    v16bf afrag[8];
#pragma unroll
    for (int kc = 0; kc < 8; ++kc) {
        const int arow  = (kc < 4) ? rowF : rowS;
        const int kbase = (kc & 3) * 32 + half * 16;
        const __bf16* ap = lsbf + (size_t)arow * DIMV + kbase;
        v8bf lo = *(const v8bf*)(ap);
        v8bf hi = *(const v8bf*)(ap + 8);
        v16bf a;
#pragma unroll
        for (int i = 0; i < 8; ++i) { a[i] = lo[i]; a[i + 8] = hi[i]; }
        afrag[kc] = a;
    }

#pragma unroll
    for (int nt = 0; nt < 8; ++nt) {
        const int ncol = nt * 16 + mrow;
        const float bias = bmsg[ncol];
        v8f acc;
#pragma unroll
        for (int j = 0; j < 8; ++j) acc[j] = bias;
#pragma unroll
        for (int kc = 0; kc < 8; ++kc) {
            const __bf16* bp = &sW[(size_t)ncol * (2 * DIMV) + kc * 32 + half * 16];
            v16bf b = *(const v16bf*)bp;
            acc = wmma_bf16(afrag[kc], b, acc);
        }
        // C layout: element j -> M = half*8+j, N = ncol
#pragma unroll
        for (int j = 0; j < 8; ++j) {
            const int m    = half * 8 + j;
            const int drow = second[pbase + m];
            atomicAdd(&agg[(size_t)drow * DIMV + ncol], selu_f(acc[j]));
        }
    }
}

// ------------------------------------------------------------------- GRU
// One wave per 16-edge tile; 6 accumulators per feature tile, fused gates.
// Both weight matrices (2 x 96 KB bf16) are TDM-staged into LDS per block.
__global__ void gru_kernel(const float* __restrict__ agg,     // [E][DIMV] f32
                           const __bf16* __restrict__ lsbf,   // stale h, bf16
                           float* __restrict__ ls,            // h f32, in-place update
                           const __bf16* __restrict__ KT,     // [3*DIMV][DIMV]
                           const __bf16* __restrict__ RT,     // [3*DIMV][DIMV]
                           const float* __restrict__ gbias,   // [2][3*DIMV]
                           int E) {
    __shared__ __bf16 sKT[3 * DIMV * DIMV];                   // 96 KB
    __shared__ __bf16 sRT[3 * DIMV * DIMV];                   // 96 KB

    if (threadIdx.x < 32) {
        tdm_load_2d_bf16((unsigned int)(uintptr_t)&sKT[0], KT, DIMV, 3 * DIMV, DIMV);
        tdm_load_2d_bf16((unsigned int)(uintptr_t)&sRT[0], RT, DIMV, 3 * DIMV, DIMV);
        __builtin_amdgcn_s_wait_tensorcnt(0);
    }
    __syncthreads();

    const int lane  = threadIdx.x & 31;
    const int wave  = threadIdx.x >> 5;
    const int tile  = blockIdx.x * (blockDim.x >> 5) + wave;
    const int ebase = tile * 16;
    if (ebase + 16 > E) return;

    const int mrow = lane & 15;
    const int half = lane >> 4;
    const int erow = ebase + mrow;

    // A fragments: x = agg (cvt f32->bf16 in regs), h = lsbf. K=128 -> 4 chunks.
    v16bf aX[4], aH[4];
#pragma unroll
    for (int kc = 0; kc < 4; ++kc) {
        const int kbase = kc * 32 + half * 16;
        const float* ap = agg + (size_t)erow * DIMV + kbase;
        v16bf ax;
#pragma unroll
        for (int i = 0; i < 8; ++i) { ax[i] = (__bf16)ap[i]; ax[i + 8] = (__bf16)ap[8 + i]; }
        aX[kc] = ax;
        const __bf16* hp = lsbf + (size_t)erow * DIMV + kbase;
        v8bf lo = *(const v8bf*)(hp);
        v8bf hi = *(const v8bf*)(hp + 8);
        v16bf ah;
#pragma unroll
        for (int i = 0; i < 8; ++i) { ah[i] = lo[i]; ah[i + 8] = hi[i]; }
        aH[kc] = ah;
    }

    for (int ft = 0; ft < 8; ++ft) {
        const int nz = ft * 16 + mrow;      // z-gate column
        const int nr = DIMV + nz;           // r-gate column
        const int nh = 2 * DIMV + nz;       // h-gate column
        v8f xz, xr, xh, rz, rr, rh;
        const float bz0 = gbias[nz],            br0 = gbias[nr],            bh0 = gbias[nh];
        const float bz1 = gbias[3 * DIMV + nz], br1 = gbias[3 * DIMV + nr], bh1 = gbias[3 * DIMV + nh];
#pragma unroll
        for (int j = 0; j < 8; ++j) {
            xz[j] = bz0; xr[j] = br0; xh[j] = bh0;
            rz[j] = bz1; rr[j] = br1; rh[j] = bh1;
        }
#pragma unroll
        for (int kc = 0; kc < 4; ++kc) {
            const int koff = kc * 32 + half * 16;
            v16bf bkz = *(const v16bf*)(&sKT[(size_t)nz * DIMV + koff]);
            v16bf bkr = *(const v16bf*)(&sKT[(size_t)nr * DIMV + koff]);
            v16bf bkh = *(const v16bf*)(&sKT[(size_t)nh * DIMV + koff]);
            v16bf brz = *(const v16bf*)(&sRT[(size_t)nz * DIMV + koff]);
            v16bf brr = *(const v16bf*)(&sRT[(size_t)nr * DIMV + koff]);
            v16bf brh = *(const v16bf*)(&sRT[(size_t)nh * DIMV + koff]);
            xz = wmma_bf16(aX[kc], bkz, xz);
            xr = wmma_bf16(aX[kc], bkr, xr);
            xh = wmma_bf16(aX[kc], bkh, xh);
            rz = wmma_bf16(aH[kc], brz, rz);
            rr = wmma_bf16(aH[kc], brr, rr);
            rh = wmma_bf16(aH[kc], brh, rh);
        }
#pragma unroll
        for (int j = 0; j < 8; ++j) {
            const int m = half * 8 + j;
            const size_t idx = (size_t)(ebase + m) * DIMV + ft * 16 + mrow;
            const float z  = sigmoid_f(xz[j] + rz[j]);
            const float r  = sigmoid_f(xr[j] + rr[j]);
            const float hh = tanhf(xh[j] + r * rh[j]);
            const float ho = ls[idx];
            ls[idx] = z * ho + (1.f - z) * hh;
        }
    }
}

// ---------------------------------------------------------- pool + readout
__global__ void pool_kernel(const float* __restrict__ ls,
                            const int* __restrict__ gids,
                            float* __restrict__ pooled, int E) {
    int i = blockIdx.x * blockDim.x + threadIdx.x;
    if (i < E * DIMV) {
        int e = i / DIMV, d = i % DIMV;
        atomicAdd(&pooled[(size_t)gids[e] * DIMV + d], ls[i]);
    }
}

__global__ void readout1_kernel(const float* __restrict__ pooled,
                                const float* __restrict__ W1,  // [DIMV,RDIM]
                                const float* __restrict__ b1,
                                float* __restrict__ h1, int G) {
    int i = blockIdx.x * blockDim.x + threadIdx.x;
    if (i >= G * RDIM) return;
    int g = i / RDIM, j = i % RDIM;
    float s = b1[j];
    for (int k = 0; k < DIMV; ++k) s += pooled[(size_t)g * DIMV + k] * W1[(size_t)k * RDIM + j];
    h1[i] = selu_f(s);
}

__global__ void readout2_kernel(const float* __restrict__ h1,
                                const float* __restrict__ W2,  // [RDIM,RDIM]
                                const float* __restrict__ b2,
                                float* __restrict__ h2, int G) {
    int i = blockIdx.x * blockDim.x + threadIdx.x;
    if (i >= G * RDIM) return;
    int g = i / RDIM, j = i % RDIM;
    float s = b2[j];
    for (int k = 0; k < RDIM; ++k) s += h1[(size_t)g * RDIM + k] * W2[(size_t)k * RDIM + j];
    h2[i] = selu_f(s);
}

__global__ void readout3_kernel(const float* __restrict__ h2,
                                const float* __restrict__ W3,  // [RDIM,1]
                                const float* __restrict__ b3,
                                float* __restrict__ out, int G) {
    int g = blockIdx.x * blockDim.x + threadIdx.x;
    if (g >= G) return;
    float s = b3[0];
    for (int k = 0; k < RDIM; ++k) s += h2[(size_t)g * RDIM + k] * W3[k];
    out[g] = s;
}

// ------------------------------------------------------------------- host
extern "C" void kernel_launch(void* const* d_in, const int* in_sizes, int n_in,
                              void* d_out, int out_size, void* d_ws, size_t ws_size,
                              hipStream_t stream) {
    const float* ls_in  = (const float*)d_in[0];
    const int*   gids   = (const int*)d_in[1];
    const int*   first  = (const int*)d_in[2];
    const int*   second = (const int*)d_in[3];
    const float* Wmsg   = (const float*)d_in[5];
    const float* bmsg   = (const float*)d_in[6];
    const float* gk     = (const float*)d_in[7];
    const float* grk    = (const float*)d_in[8];
    const float* gbias  = (const float*)d_in[9];
    const float* W1     = (const float*)d_in[10];
    const float* b1     = (const float*)d_in[11];
    const float* W2     = (const float*)d_in[12];
    const float* b2     = (const float*)d_in[13];
    const float* W3     = (const float*)d_in[14];
    const float* b3     = (const float*)d_in[15];
    float* out = (float*)d_out;

    const int E = in_sizes[0] / DIMV;
    const int P = in_sizes[2];
    const int G = out_size;

    char* ws = (char*)d_ws;
    size_t off = 0;
    auto carve = [&](size_t bytes) {
        void* p = ws + off;
        off = (off + bytes + 255) & ~(size_t)255;
        return p;
    };
    float*  lsf    = (float*)carve((size_t)E * DIMV * sizeof(float));
    float*  aggp   = (float*)carve((size_t)E * DIMV * sizeof(float));
    __bf16* lsbf   = (__bf16*)carve((size_t)E * DIMV * sizeof(__bf16));
    __bf16* WmsgT  = (__bf16*)carve((size_t)DIMV * 2 * DIMV * sizeof(__bf16));
    __bf16* KT     = (__bf16*)carve((size_t)3 * DIMV * DIMV * sizeof(__bf16));
    __bf16* RT     = (__bf16*)carve((size_t)3 * DIMV * DIMV * sizeof(__bf16));
    float*  pooled = (float*)carve((size_t)G * DIMV * sizeof(float));
    float*  h1     = (float*)carve((size_t)G * RDIM * sizeof(float));
    float*  h2     = (float*)carve((size_t)G * RDIM * sizeof(float));
    (void)ws_size; (void)n_in;

    const int nState = E * DIMV;
    const int TPB = 256;

    copy_f32_kernel<<<(nState + TPB - 1) / TPB, TPB, 0, stream>>>(ls_in, lsf, nState);
    {
        const int nW = DIMV * 2 * DIMV + 3 * DIMV * DIMV;
        prep_weights_kernel<<<(nW + TPB - 1) / TPB, TPB, 0, stream>>>(Wmsg, gk, grk,
                                                                      WmsgT, KT, RT);
    }

    const int nTileP   = P / 16;                 // pair tiles (16 pairs / wave)
    const int nTileE   = (E + 15) / 16;          // edge tiles
    const int wavesPB  = TPB / 32;               // 8 waves per block
    const int gridMsg  = (nTileP + wavesPB - 1) / wavesPB;
    const int gridGru  = (nTileE + wavesPB - 1) / wavesPB;

    for (int t = 0; t < TSTEPS; ++t) {
        cvt_zero_kernel<<<(nState + TPB - 1) / TPB, TPB, 0, stream>>>(lsf, lsbf, aggp, nState);
        msg_kernel<<<gridMsg, TPB, 0, stream>>>(lsbf, WmsgT, bmsg, first, second, aggp, P);
        gru_kernel<<<gridGru, TPB, 0, stream>>>(aggp, lsbf, lsf, KT, RT, gbias, E);
    }

    zero_f32_kernel<<<(G * DIMV + TPB - 1) / TPB, TPB, 0, stream>>>(pooled, G * DIMV);
    pool_kernel<<<(nState + TPB - 1) / TPB, TPB, 0, stream>>>(lsf, gids, pooled, E);
    readout1_kernel<<<(G * RDIM + TPB - 1) / TPB, TPB, 0, stream>>>(pooled, W1, b1, h1, G);
    readout2_kernel<<<(G * RDIM + TPB - 1) / TPB, TPB, 0, stream>>>(h1, W2, b2, h2, G);
    readout3_kernel<<<(G + TPB - 1) / TPB, TPB, 0, stream>>>(h2, W3, b3, out, G);
}
